// GCN_76751065580100
// MI455X (gfx1250) — matrix operand order repaired and checked by reading, not compiled, not run on hardware
//
#include <hip/hip_runtime.h>

// ---------- types for CDNA5 WMMA ----------
typedef __attribute__((ext_vector_type(16))) __bf16 bf16x16;
typedef __attribute__((ext_vector_type(8)))  __bf16 bf16x8;
typedef __attribute__((ext_vector_type(4)))  __bf16 bf16x4;
typedef __attribute__((ext_vector_type(8)))  float  f32x8;

static __device__ __forceinline__ __bf16 f2bf(float f) {
  // round-to-nearest-even fp32 -> bf16
  unsigned u = __builtin_bit_cast(unsigned, f);
  u += 0x7FFFu + ((u >> 16) & 1u);
  unsigned short h = (unsigned short)(u >> 16);
  return __builtin_bit_cast(__bf16, h);
}

#define TM 128
#define TN 128
#define TK 32

// C[M,N] = A[M,K] @ B[K,N], fp32 in/out, bf16 V_WMMA_F32_16X16X32_BF16 math.
// Block: 256 threads = 8 waves; wave tile 32x64 (2x4 WMMA tiles).
// Software-pipelined: register-staged global loads for tile k+1 overlap the
// WMMA compute of tile k; double-buffered LDS -> one barrier per K-step.
__global__ __launch_bounds__(256) void gemm_bf16_wmma(
    const float* __restrict__ A, const float* __restrict__ B,
    float* __restrict__ C, int K, int lda, int ldb, int ldc)
{
  __shared__ __bf16 As[2][TM * TK];   // [row][k], k-contiguous
  __shared__ __bf16 Bs[2][TN * TK];   // [col][k], transposed: frags k-contiguous

  const int tid  = threadIdx.x;
  const int lane = tid & 31;
  const int wave = tid >> 5;
  const int wm   = wave & 3;       // M strip of 32
  const int wn   = wave >> 2;      // N strip of 64
  const int l16  = lane & 15;
  const int lhi  = lane >> 4;

  const int mb = blockIdx.x * TM;
  const int nb = blockIdx.y * TN;

  f32x8 acc[2][4];
  #pragma unroll
  for (int mt = 0; mt < 2; ++mt)
    #pragma unroll
    for (int nt = 0; nt < 4; ++nt)
      #pragma unroll
      for (int j = 0; j < 8; ++j) acc[mt][nt][j] = 0.0f;

  const int a_cg = (tid & 7) * 4;      // A: col group within 32
  const int a_r0 = tid >> 3;           // A: row 0..31 (+32 per rep)
  const int b_ng = (tid & 31) * 4;     // B: col group within 128
  const int b_k0 = tid >> 5;           // B: k 0..7 (+8 per rep)

  float4 areg[4], breg[4];

  // ---- prologue: fetch tile 0 into registers ----
  #pragma unroll
  for (int i = 0; i < 4; ++i)
    areg[i] = *(const float4*)(A + (size_t)(mb + a_r0 + i * 32) * lda + a_cg);
  #pragma unroll
  for (int i = 0; i < 4; ++i)
    breg[i] = *(const float4*)(B + (size_t)(b_k0 + i * 8) * ldb + nb + b_ng);

  // convert + store tile 0 into LDS buffer 0
  #pragma unroll
  for (int i = 0; i < 4; ++i) {
    const int r = a_r0 + i * 32;
    bf16x4 h; h.x = f2bf(areg[i].x); h.y = f2bf(areg[i].y);
    h.z = f2bf(areg[i].z); h.w = f2bf(areg[i].w);
    *(bf16x4*)(&As[0][r * TK + a_cg]) = h;
  }
  #pragma unroll
  for (int i = 0; i < 4; ++i) {
    const int kk = b_k0 + i * 8;
    Bs[0][(b_ng + 0) * TK + kk] = f2bf(breg[i].x);
    Bs[0][(b_ng + 1) * TK + kk] = f2bf(breg[i].y);
    Bs[0][(b_ng + 2) * TK + kk] = f2bf(breg[i].z);
    Bs[0][(b_ng + 3) * TK + kk] = f2bf(breg[i].w);
  }

  int buf = 0;
  for (int k0 = 0; k0 < K; k0 += TK) {
    __syncthreads();
    const int knext = k0 + TK;

    // ---- issue global loads for the NEXT tile (in flight during compute) ----
    if (knext < K) {
      #pragma unroll
      for (int i = 0; i < 4; ++i)
        areg[i] = *(const float4*)(A + (size_t)(mb + a_r0 + i * 32) * lda + knext + a_cg);
      #pragma unroll
      for (int i = 0; i < 4; ++i)
        breg[i] = *(const float4*)(B + (size_t)(knext + b_k0 + i * 8) * ldb + nb + b_ng);
    }

    // ---- compute current tile from LDS[buf] ----
    // A 16x32 bf16 frag: lanes 0-15 hold K[0..7]+K[16..23]; lanes 16-31 K[8..15]+K[24..31]
    bf16x16 afrag[2];
    #pragma unroll
    for (int mt = 0; mt < 2; ++mt) {
      const int row = wm * 32 + mt * 16 + l16;
      const int c = lhi * 8;
      union { bf16x16 v; bf16x8 h[2]; } u;
      u.h[0] = *(const bf16x8*)(&As[buf][row * TK + c]);
      u.h[1] = *(const bf16x8*)(&As[buf][row * TK + c + 16]);
      afrag[mt] = u.v;
    }
    // B 32x16 bf16 frag: lanes 0-15 hold K[0..7]+K[16..23]; lanes 16-31 K[8..15]+K[24..31], col = lane&15
    bf16x16 bfrag[4];
    #pragma unroll
    for (int nt = 0; nt < 4; ++nt) {
      const int coln = wn * 64 + nt * 16 + l16;
      union { bf16x16 v; bf16x8 h[2]; } ub;
      ub.h[0] = *(const bf16x8*)(&Bs[buf][coln * TK + lhi * 8]);
      ub.h[1] = *(const bf16x8*)(&Bs[buf][coln * TK + lhi * 8 + 16]);
      bfrag[nt] = ub.v;
    }
    #pragma unroll
    for (int mt = 0; mt < 2; ++mt)
      #pragma unroll
      for (int nt = 0; nt < 4; ++nt)
        acc[mt][nt] = __builtin_amdgcn_wmma_f32_16x16x32_bf16(
            false, afrag[mt], false, bfrag[nt], (short)0, acc[mt][nt],
            false, false);

    // ---- convert + store next tile into the alternate buffer ----
    // Safe with one barrier/iter: buf^1 was last READ before this iteration's
    // barrier, and this store happens after it.
    if (knext < K) {
      const int nbuf = buf ^ 1;
      #pragma unroll
      for (int i = 0; i < 4; ++i) {
        const int r = a_r0 + i * 32;
        bf16x4 h; h.x = f2bf(areg[i].x); h.y = f2bf(areg[i].y);
        h.z = f2bf(areg[i].z); h.w = f2bf(areg[i].w);
        *(bf16x4*)(&As[nbuf][r * TK + a_cg]) = h;
      }
      #pragma unroll
      for (int i = 0; i < 4; ++i) {
        const int kk = b_k0 + i * 8;
        Bs[nbuf][(b_ng + 0) * TK + kk] = f2bf(breg[i].x);
        Bs[nbuf][(b_ng + 1) * TK + kk] = f2bf(breg[i].y);
        Bs[nbuf][(b_ng + 2) * TK + kk] = f2bf(breg[i].z);
        Bs[nbuf][(b_ng + 3) * TK + kk] = f2bf(breg[i].w);
      }
    }
    buf ^= 1;
  }

  // ---- store C: VGPR r <-> M = r + 8*(lane>=16), N = lane&15 ----
  #pragma unroll
  for (int mt = 0; mt < 2; ++mt)
    #pragma unroll
    for (int nt = 0; nt < 4; ++nt) {
      const int gcol = nb + wn * 64 + nt * 16 + l16;
      #pragma unroll
      for (int r = 0; r < 8; ++r) {
        const int grow = mb + wm * 32 + mt * 16 + lhi * 8 + r;
        C[(size_t)grow * ldc + gcol] = acc[mt][nt][r];
      }
    }
}

// ---------- GCN glue kernels ----------
__global__ void k_zero(float* p, int n) {
  int i = blockIdx.x * 256 + threadIdx.x; if (i < n) p[i] = 0.0f;
}
__global__ void k_fill1(float* p, int n) {
  int i = blockIdx.x * 256 + threadIdx.x; if (i < n) p[i] = 1.0f;
}
__global__ void k_deg(const int* __restrict__ ei, float* __restrict__ deg, int E) {
  int e = blockIdx.x * 256 + threadIdx.x;
  if (e < E) atomicAdd(&deg[ei[E + e]], 1.0f);  // in-degree over dst
}
__global__ void k_rsqrt(const float* __restrict__ deg, float* __restrict__ dinv, int n) {
  int i = blockIdx.x * 256 + threadIdx.x;
  if (i < n) dinv[i] = rsqrtf(deg[i]);          // deg >= 1 (self-loop) always
}
// out[dst] += dinv[src]*dinv[dst] * h[src]  (edge messages)
__global__ void k_aggregate(const float* __restrict__ h, const int* __restrict__ ei,
                            const float* __restrict__ dinv, float* __restrict__ out,
                            int E, int F) {
  const int e = blockIdx.x;
  const int f = blockIdx.y * 256 + threadIdx.x;
  const int s = ei[e], d = ei[E + e];
  const float nm = dinv[s] * dinv[d];
  atomicAdd(&out[(size_t)d * F + f], nm * h[(size_t)s * F + f]);
}
// agg[i] = relu(agg[i] + dinv[i]^2 * h[i] + b)   (self-loop + bias + relu, in place)
__global__ void k_self_bias_relu(float* __restrict__ agg, const float* __restrict__ h,
                                 const float* __restrict__ dinv, const float* __restrict__ b,
                                 int F) {
  const int i = blockIdx.x;
  const int f = blockIdx.y * 256 + threadIdx.x;
  const float di = dinv[i];
  const float v = agg[(size_t)i * F + f] + di * di * h[(size_t)i * F + f] + b[f];
  agg[(size_t)i * F + f] = fmaxf(v, 0.0f);
}
// mean over nodes -> fc[256,2] -> log_softmax -> out[2]
__global__ void k_head(const float* __restrict__ h2, const float* __restrict__ fcW,
                       const float* __restrict__ fcb, float* __restrict__ out) {
  __shared__ float g[256];
  const int c = threadIdx.x;
  float s = 0.0f;
  for (int r = 0; r < 2048; ++r) s += h2[r * 256 + c];
  g[c] = s * (1.0f / 2048.0f);
  __syncthreads();
  if (c == 0) {
    float l0 = fcb[0], l1 = fcb[1];
    for (int j = 0; j < 256; ++j) { l0 += g[j] * fcW[2 * j]; l1 += g[j] * fcW[2 * j + 1]; }
    const float m = fmaxf(l0, l1);
    const float lse = m + logf(expf(l0 - m) + expf(l1 - m));
    out[0] = l0 - lse;
    out[1] = l1 - lse;
  }
}

extern "C" void kernel_launch(void* const* d_in, const int* in_sizes, int n_in,
                              void* d_out, int out_size, void* d_ws, size_t ws_size,
                              hipStream_t stream) {
  (void)n_in; (void)out_size; (void)ws_size;
  const float* x   = (const float*)d_in[0];
  const int*   ei  = (const int*)  d_in[1];
  const float* W1  = (const float*)d_in[2];
  const float* b1  = (const float*)d_in[3];
  const float* W2  = (const float*)d_in[4];
  const float* b2  = (const float*)d_in[5];
  const float* fcW = (const float*)d_in[6];
  const float* fcb = (const float*)d_in[7];
  float* out = (float*)d_out;

  const int Nn = 2048, F0 = 50176, F1 = 1024, F2 = 256;
  const int E = in_sizes[1] / 2;   // 32768

  float* ws   = (float*)d_ws;
  float* xw1  = ws;                         // [Nn, F1]  X @ W1
  float* h1   = xw1 + (size_t)Nn * F1;      // [Nn, F1]  aggregated + relu
  float* xw2  = h1  + (size_t)Nn * F1;      // [Nn, F2]  H1 @ W2
  float* h2   = xw2 + (size_t)Nn * F2;      // [Nn, F2]  aggregated + relu
  float* deg  = h2  + (size_t)Nn * F2;      // [Nn]
  float* dinv = deg + Nn;                   // [Nn]

  // degrees (self-loops) + zero the aggregation accumulators each call
  k_fill1<<<(Nn + 255) / 256, 256, 0, stream>>>(deg, Nn);
  k_zero <<<(Nn * F1 + 255) / 256, 256, 0, stream>>>(h1, Nn * F1);
  k_zero <<<(Nn * F2 + 255) / 256, 256, 0, stream>>>(h2, Nn * F2);
  k_deg  <<<(E + 255) / 256, 256, 0, stream>>>(ei, deg, E);
  k_rsqrt<<<(Nn + 255) / 256, 256, 0, stream>>>(deg, dinv, Nn);

  // layer 1: 2048x50176 @ 50176x1024 (the 210-GFLOP hot spot, bf16 WMMA)
  gemm_bf16_wmma<<<dim3(Nn / TM, F1 / TN), 256, 0, stream>>>(x, W1, xw1, F0, F0, F1, F1);
  k_aggregate<<<dim3(E, F1 / 256), 256, 0, stream>>>(xw1, ei, dinv, h1, E, F1);
  k_self_bias_relu<<<dim3(Nn, F1 / 256), 256, 0, stream>>>(h1, xw1, dinv, b1, F1);

  // layer 2: 2048x1024 @ 1024x256
  gemm_bf16_wmma<<<dim3(Nn / TM, F2 / TN), 256, 0, stream>>>(h1, W2, xw2, F1, F1, F2, F2);
  k_aggregate<<<dim3(E, F2 / 256), 256, 0, stream>>>(xw2, ei, dinv, h2, E, F2);
  k_self_bias_relu<<<dim3(Nn, F2 / 256), 256, 0, stream>>>(h2, xw2, dinv, b2, F2);

  // mean pool + fc + log_softmax
  k_head<<<1, 256, 0, stream>>>(h2, fcW, fcb, out);
}
